// WARVD_12730283065776
// MI455X (gfx1250) — matrix operand order repaired and checked by reading, not compiled, not run on hardware
//
#include <hip/hip_runtime.h>

typedef __attribute__((ext_vector_type(16))) __bf16 v16bf;
typedef __attribute__((ext_vector_type(8)))  float  v8f;

static constexpr int NN  = 50000;   // nodes
static constexpr int NE  = 800000;  // edges
static constexpr int INC = 100;     // input channels
static constexpr int H   = 128;     // hidden
static constexpr int NG  = 64;      // graphs

__device__ __forceinline__ float geluf(float x){
  return 0.5f * x * (1.0f + erff(x * 0.70710678118654752440f));
}
__device__ __forceinline__ float sigmf(float x){
  return 1.0f / (1.0f + expf(-x));
}

// Native CDNA5 no-return f32 atomic add at device scope (STOREcnt-tracked).
// Avoids any possibility of a CAS-loop lowering for atomicAdd(float*).
__device__ __forceinline__ void atomAddF32(float* p, float v){
  asm volatile("global_atomic_add_f32 %0, %1, off scope:SCOPE_DEV"
               : : "v"(p), "v"(v) : "memory");
}

// ---------------- WMMA helpers: bf16 16x16x32, f32 accumulate ----------------
// A fragment from LDS tile (row-major [16][lda] bf16):
//   lane<16:  M=lane,    halves h0..7 = K[kb..kb+7],   h8..15 = K[kb+16..kb+23]
//   lane>=16: M=lane-16, halves h0..7 = K[kb+8..kb+15],h8..15 = K[kb+24..kb+31]
__device__ __forceinline__ v16bf ldA(const __bf16* A, int lda, int kb, int lane){
  int row  = lane & 15;
  int koff = kb + ((lane & 16) ? 8 : 0);
  const __bf16* p = A + row * lda + koff;
  v16bf f;
#pragma unroll
  for (int i = 0; i < 8; ++i){ f[i] = p[i]; f[8 + i] = p[16 + i]; }
  return f;
}
// B fragment from global Bt stored as [N][K] row-major bf16 (i.e. B transposed):
//   lane<16:  N=n0+lane,    K=kb..kb+7,kb+16..kb+23 ;  lane>=16: N=n0+lane-16, K=kb+8..kb+15,kb+24..kb+31
__device__ __forceinline__ v16bf ldB(const __bf16* Bt, int ldb, int n0, int kb, int lane){
  int n  = n0 + (lane & 15);
  int kk = kb + ((lane & 16) ? 8 : 0);
  const __bf16* p = Bt + (size_t)n * ldb + kk;
  v16bf f;
#pragma unroll
  for (int i = 0; i < 8; ++i){ f[i] = p[i]; f[8 + i] = p[16 + i]; }
  return f;
}
#define WMMA_BF16(a, b, c) \
  __builtin_amdgcn_wmma_f32_16x16x32_bf16(false, (a), false, (b), (short)0, (c), false, false)

// ---------------- small utility kernels ----------------
__global__ void k_zero(float* p, int n){
  int i = blockIdx.x * 256 + threadIdx.x;
  if (i < n) p[i] = 0.0f;
}
__global__ void k_cnt(const int* __restrict__ idx, float* out, int n){
  int i = blockIdx.x * 256 + threadIdx.x;
  if (i < n) atomAddF32(&out[idx[i]], 1.0f);
}

// edge-attr tables: tab[v] = gelu(emb[v]@W_ee + b_ee); tab2[v] = tab[v]@Wa1[256:384] + ba1
__global__ __launch_bounds__(128) void k_tables(const float* __restrict__ emb,
    const float* __restrict__ W_ee, const float* __restrict__ b_ee,
    const float* __restrict__ Wa1, const float* __restrict__ ba1,
    float* tab, float* tab2){
  __shared__ float TL[16][128];
  int j = threadIdx.x;
  for (int v = 0; v < 16; ++v){
    float a = b_ee[j];
    for (int c = 0; c < 32; ++c) a += emb[v * 32 + c] * W_ee[c * H + j];
    float g = geluf(a);
    TL[v][j] = g; tab[v * H + j] = g;
  }
  __syncthreads();
  for (int v = 0; v < 16; ++v){
    float a = ba1[j];
    for (int c = 0; c < H; ++c) a += TL[v][c] * Wa1[(256 + c) * H + j];
    tab2[v * H + j] = a;
  }
}

// W_in^T (K padded 100->128 with zeros), bf16, layout [n=128][k=128]
__global__ void k_cv_wint(const float* __restrict__ Win, __bf16* o){
  int i = blockIdx.x * 256 + threadIdx.x;
  if (i < 128 * 128){
    int n = i >> 7, k = i & 127;
    o[i] = (__bf16)((k < INC) ? Win[k * H + n] : 0.0f);
  }
}
// [Wa1a|Wa1b]^T bf16, layout [n=256][k=128]
__global__ void k_cv_btp(const float* __restrict__ Wa1, __bf16* o){
  int i = blockIdx.x * 256 + threadIdx.x;
  if (i < 256 * 128){
    int n = i >> 7, k = i & 127;
    float v = (n < 128) ? Wa1[k * H + n] : Wa1[(128 + k) * H + (n - 128)];
    o[i] = (__bf16)v;
  }
}
__global__ void k_cv_f2b(const float* __restrict__ in, __bf16* o, int n){
  int i = blockIdx.x * 256 + threadIdx.x;
  if (i < n) o[i] = (__bf16)in[i];
}
// Combined weight Gt[l,k][n][k_] = sum_j ggc_w[l,k][k_][j] * gru_wih[l][n][j]  (bf16)
__global__ __launch_bounds__(128) void k_gt(const float* __restrict__ ggc,
    const float* __restrict__ wih, __bf16* Gt){
  __shared__ float wr[128];
  int lk = blockIdx.x / 384, n = blockIdx.x % 384;
  int l = lk / 3;
  int t = threadIdx.x;
  wr[t] = wih[((size_t)l * 384 + n) * H + t];
  __syncthreads();
  const float* Wm = ggc + (size_t)lk * H * H;
  float acc = 0.0f;
  for (int j = 0; j < H; ++j) acc += Wm[t * H + j] * wr[j];
  Gt[((size_t)lk * 384 + n) * H + t] = (__bf16)acc;
}

// ---------------- x0 = x @ W_in + b_in ; also h = x0 (f32 + bf16 shadow) ----------------
__global__ __launch_bounds__(128) void k_x0(const float* __restrict__ x,
    const __bf16* __restrict__ WinT, const float* __restrict__ b_in,
    float* X0, float* Hf, __bf16* Hb){
  __shared__ __bf16 xA[16][128];
  __shared__ float  OT[16][128];
  int tid = threadIdx.x, wv = tid >> 5, lane = tid & 31;
  int g0 = blockIdx.x * 16;
  for (int i = tid; i < 16 * 128; i += 128){
    int r = i >> 7, c = i & 127;
    xA[r][c] = (__bf16)((c < INC) ? x[(size_t)(g0 + r) * INC + c] : 0.0f);
  }
  __syncthreads();
#pragma unroll
  for (int tt = 0; tt < 2; ++tt){
    int n0 = (wv * 2 + tt) * 16;
    v8f c = {};
#pragma unroll
    for (int kb = 0; kb < 128; kb += 32){
      v16bf a = ldA(&xA[0][0], 128, kb, lane);
      v16bf b = ldB(WinT, 128, n0, kb, lane);
      c = WMMA_BF16(a, b, c);
    }
    int col = lane & 15, rb = (lane & 16) ? 8 : 0;
#pragma unroll
    for (int r = 0; r < 8; ++r) OT[rb + r][n0 + col] = c[r] + b_in[n0 + col];
  }
  __syncthreads();
  for (int i = tid; i < 16 * 128; i += 128){
    int r = i >> 7, cc = i & 127;
    float v = OT[r][cc];
    size_t o = (size_t)(g0 + r) * H + cc;
    X0[o] = v; Hf[o] = v; Hb[o] = (__bf16)v;
  }
}

// ---------------- scatter segment-sum of node rows over dst ----------------
__global__ __launch_bounds__(256) void k_scat(const float* __restrict__ rows,
    const int* __restrict__ src, const int* __restrict__ dst, float* acc){
  int e = blockIdx.x * 8 + (threadIdx.x >> 5);
  int lane = threadIdx.x & 31;
  int s = src[e], d = dst[e];
  const float4* rp = (const float4*)(rows + (size_t)s * H);
  float4 v = rp[lane];
  float* ap = acc + (size_t)d * H + lane * 4;
  atomAddF32(ap + 0, v.x); atomAddF32(ap + 1, v.y);
  atomAddF32(ap + 2, v.z); atomAddF32(ap + 3, v.w);
}

// ---------------- fused GRU step: gi = (sacc/deg)@Gt + bih ; gh = hl@whh^T + bhh ; gates ----------------
__global__ __launch_bounds__(256) void k_gru(const float* __restrict__ sacc,
    const float* __restrict__ deg,
    const __bf16* __restrict__ hbf_in, const float* __restrict__ hf_in,
    const __bf16* __restrict__ GtB, const __bf16* __restrict__ whhB,
    const float* __restrict__ bihl, const float* __restrict__ bhhl,
    float* hlf, __bf16* hlb){
  __shared__ __bf16 sA[16][128];
  __shared__ __bf16 hA[16][128];
  __shared__ float  GI[16][384];
  __shared__ float  GH[16][384];
  int tid = threadIdx.x, wv = tid >> 5, lane = tid & 31;
  int g0 = blockIdx.x * 16;
  for (int i = tid; i < 16 * 128; i += 256){
    int r = i >> 7, c = i & 127;
    size_t o = (size_t)(g0 + r) * H + c;
    float dv = fmaxf(deg[g0 + r], 1.0f);
    sA[r][c] = (__bf16)(sacc[o] / dv);
    hA[r][c] = hbf_in[o];
  }
  __syncthreads();
#pragma unroll
  for (int tt = 0; tt < 6; ++tt){
    int t = wv * 6 + tt;            // 0..47 : 24 gi tiles then 24 gh tiles
    bool isGH = (t >= 24);
    int n0 = (isGH ? (t - 24) : t) * 16;
    const __bf16* A  = isGH ? &hA[0][0] : &sA[0][0];
    const __bf16* Bt = isGH ? whhB : GtB;
    v8f c = {};
#pragma unroll
    for (int kb = 0; kb < 128; kb += 32){
      v16bf a = ldA(A, 128, kb, lane);
      v16bf b = ldB(Bt, 128, n0, kb, lane);
      c = WMMA_BF16(a, b, c);
    }
    int col = lane & 15, rb = (lane & 16) ? 8 : 0;
    const float* bias = isGH ? bhhl : bihl;
    float* O = isGH ? &GH[0][0] : &GI[0][0];
#pragma unroll
    for (int r = 0; r < 8; ++r) O[(rb + r) * 384 + n0 + col] = c[r] + bias[n0 + col];
  }
  __syncthreads();
  for (int i = tid; i < 16 * 128; i += 256){
    int r = i >> 7, j = i & 127;
    size_t o = (size_t)(g0 + r) * H + j;
    float rr = sigmf(GI[r][j]       + GH[r][j]);
    float zz = sigmf(GI[r][128 + j] + GH[r][128 + j]);
    float nn = tanhf(GI[r][256 + j] + rr * GH[r][256 + j]);
    float hv = hf_in[o];
    float hn = (1.0f - zz) * nn + zz * hv;
    hlf[o] = hn; hlb[o] = (__bf16)hn;
  }
}

// ---------------- P = h @ [Wa1a|Wa1b]  ->  [N,256] ----------------
__global__ __launch_bounds__(128) void k_p12(const __bf16* __restrict__ hbf,
    const __bf16* __restrict__ BtP, float* P){
  __shared__ __bf16 hA[16][128];
  __shared__ float  OT[16][256];
  int tid = threadIdx.x, wv = tid >> 5, lane = tid & 31;
  int g0 = blockIdx.x * 16;
  for (int i = tid; i < 16 * 128; i += 128){
    int r = i >> 7, c = i & 127;
    hA[r][c] = hbf[(size_t)(g0 + r) * H + c];
  }
  __syncthreads();
#pragma unroll
  for (int tt = 0; tt < 4; ++tt){
    int n0 = (wv * 4 + tt) * 16;
    v8f c = {};
#pragma unroll
    for (int kb = 0; kb < 128; kb += 32){
      v16bf a = ldA(&hA[0][0], 128, kb, lane);
      v16bf b = ldB(BtP, 128, n0, kb, lane);
      c = WMMA_BF16(a, b, c);
    }
    int col = lane & 15, rb = (lane & 16) ? 8 : 0;
#pragma unroll
    for (int r = 0; r < 8; ++r) OT[rb + r][n0 + col] = c[r];
  }
  __syncthreads();
  for (int i = tid; i < 16 * 256; i += 128){
    int r = i >> 8, c = i & 255;
    P[(size_t)(g0 + r) * 256 + c] = OT[r][c];
  }
}

// ---------------- per-edge attention gate + weighted scatter of table rows ----------------
__global__ __launch_bounds__(256) void k_attn(const float* __restrict__ P,
    const int* __restrict__ src, const int* __restrict__ dst, const int* __restrict__ attr,
    const float* __restrict__ tab, const float* __restrict__ tab2,
    const float* __restrict__ Wa2, const float* __restrict__ ba2, float* wm){
  int e = blockIdx.x * 8 + (threadIdx.x >> 5);
  int lane = threadIdx.x & 31;
  int s = src[e], d = dst[e], av = attr[e];
  const float* p1 = P + (size_t)s * 256;
  const float* p2 = P + (size_t)d * 256 + 128;
  const float* t2 = tab2 + av * H;
  float part = 0.0f;
#pragma unroll
  for (int i = 0; i < 4; ++i){
    int j = i * 32 + lane;
    float xx = p1[j] + p2[j] + t2[j];
    part += geluf(xx) * Wa2[j];
  }
#pragma unroll
  for (int o = 16; o; o >>= 1) part += __shfl_xor(part, o, 32);
  float a = sigmf(part + ba2[0]);
  const float* tb = tab + av * H;
  float* wp = wm + (size_t)d * H;
#pragma unroll
  for (int i = 0; i < 4; ++i){
    int j = i * 32 + lane;
    atomAddF32(&wp[j], a * tb[j]);
  }
}

// ---------------- h = hl + wm/deg ----------------
__global__ void k_finish(const float* __restrict__ hl, const float* __restrict__ wm,
    const float* __restrict__ deg, float* hf, __bf16* hb){
  int i = blockIdx.x * 256 + threadIdx.x;
  int node = i >> 7;
  float dv = fmaxf(deg[node], 1.0f);
  float v = hl[i] + wm[i] / dv;
  hf[i] = v; hb[i] = (__bf16)v;
}

// ---------------- graph pooling: sum [x0|h] per graph ----------------
__global__ void k_pool(const float* __restrict__ X0, const float* __restrict__ Hf,
    const int* __restrict__ batch, float* gsum){
  int i = blockIdx.x * 256 + threadIdx.x;   // NN*256 threads
  int node = i >> 8, c = i & 255;
  float v = (c < 128) ? X0[(size_t)node * H + c] : Hf[(size_t)node * H + (c - 128)];
  atomAddF32(&gsum[batch[node] * 256 + c], v);
}

// ---------------- readout head: linear + batchnorm(train) + gelu + linear + sigmoid ----------------
__global__ __launch_bounds__(128) void k_head(const float* __restrict__ gsum,
    const float* __restrict__ bcnt, const float* __restrict__ W1, const float* __restrict__ b1,
    const float* __restrict__ gamma, const float* __restrict__ beta,
    const float* __restrict__ W2, float* out){
  __shared__ float F[64][128];
  int j = threadIdx.x;
  for (int b = 0; b < NG; ++b){
    float cnt = fmaxf(bcnt[b], 1.0f);
    float a = b1[j];
    for (int c = 0; c < 256; ++c) a += (gsum[b * 256 + c] / cnt) * W1[c * H + j];
    F[b][j] = a;
  }
  float mu = 0.0f;
  for (int b = 0; b < NG; ++b) mu += F[b][j];
  mu *= (1.0f / NG);
  float var = 0.0f;
  for (int b = 0; b < NG; ++b){ float dd = F[b][j] - mu; var += dd * dd; }
  var *= (1.0f / NG);
  float inv = rsqrtf(var + 1e-5f);
  for (int b = 0; b < NG; ++b){
    float bn = gamma[j] * (F[b][j] - mu) * inv + beta[j];
    float m = geluf(bn);
    F[b][j] = m;
    out[64 + b * H + j] = m;     // mlp output (second tuple element)
  }
  __syncthreads();
  if (j < NG){
    float s = 0.0f;
    for (int c = 0; c < H; ++c) s += F[j][c] * W2[c];
    out[j] = sigmf(s);           // first tuple element
  }
}

// ---------------- host launch ----------------
extern "C" void kernel_launch(void* const* d_in, const int* in_sizes, int n_in,
                              void* d_out, int out_size, void* d_ws, size_t ws_size,
                              hipStream_t stream){
  const float* x     = (const float*)d_in[0];
  const int*   ei    = (const int*)  d_in[1];
  const int*   eattr = (const int*)  d_in[2];
  const int*   batch = (const int*)  d_in[3];
  const float* W_in  = (const float*)d_in[4];
  const float* b_in  = (const float*)d_in[5];
  const float* emb   = (const float*)d_in[6];
  const float* W_ee  = (const float*)d_in[7];
  const float* b_ee  = (const float*)d_in[8];
  const float* ggc   = (const float*)d_in[9];
  const float* wih   = (const float*)d_in[10];
  const float* whh   = (const float*)d_in[11];
  const float* bih   = (const float*)d_in[12];
  const float* bhh   = (const float*)d_in[13];
  const float* Wa1   = (const float*)d_in[14];
  const float* ba1   = (const float*)d_in[15];
  const float* Wa2   = (const float*)d_in[16];
  const float* ba2   = (const float*)d_in[17];
  const float* W1    = (const float*)d_in[18];
  const float* b1    = (const float*)d_in[19];
  const float* gam   = (const float*)d_in[20];
  const float* bet   = (const float*)d_in[21];
  const float* W2    = (const float*)d_in[22];
  const int* srcI = ei;
  const int* dstI = ei + NE;

  char* base = (char*)d_ws;
  size_t off = 0;
  auto take = [&](size_t bytes)->char*{
    char* p = base + off;
    off = (off + bytes + 255) & ~(size_t)255;
    return p;
  };
  float*  X0   = (float*) take((size_t)NN * H * 4);
  float*  Hf   = (float*) take((size_t)NN * H * 4);
  __bf16* Hb   = (__bf16*)take((size_t)NN * H * 2);
  float*  HLf  = (float*) take((size_t)NN * H * 4);
  __bf16* HLb  = (__bf16*)take((size_t)NN * H * 2);
  float*  SACC = (float*) take((size_t)NN * H * 4);   // reused as wm accumulator
  float*  P    = (float*) take((size_t)NN * 256 * 4);
  float*  DEG  = (float*) take((size_t)NN * 4);
  float*  BCNT = (float*) take((size_t)NG * 4);
  float*  GSUM = (float*) take((size_t)NG * 256 * 4);
  float*  TAB  = (float*) take(16 * H * 4);
  float*  TAB2 = (float*) take(16 * H * 4);
  __bf16* WINT = (__bf16*)take(128 * 128 * 2);
  __bf16* BTP  = (__bf16*)take(256 * 128 * 2);
  __bf16* WHHB = (__bf16*)take((size_t)3 * 384 * 128 * 2);
  __bf16* GT   = (__bf16*)take((size_t)9 * 384 * 128 * 2);
  (void)n_in; (void)in_sizes; (void)out_size; (void)ws_size;

  // prep: counts, tables, bf16 weight copies
  k_zero<<<(NN + 255) / 256, 256, 0, stream>>>(DEG, NN);
  k_zero<<<1, 256, 0, stream>>>(BCNT, NG);
  k_zero<<<(NG * 256 + 255) / 256, 256, 0, stream>>>(GSUM, NG * 256);
  k_cnt<<<NE / 256, 256, 0, stream>>>(dstI, DEG, NE);
  k_cnt<<<(NN + 255) / 256, 256, 0, stream>>>(batch, BCNT, NN);
  k_tables<<<1, 128, 0, stream>>>(emb, W_ee, b_ee, Wa1, ba1, TAB, TAB2);
  k_cv_wint<<<64, 256, 0, stream>>>(W_in, WINT);
  k_cv_btp<<<128, 256, 0, stream>>>(Wa1, BTP);
  k_cv_f2b<<<576, 256, 0, stream>>>(whh, WHHB, 3 * 384 * 128);
  k_gt<<<9 * 384, 128, 0, stream>>>(ggc, wih, GT);

  // input projection
  k_x0<<<NN / 16, 128, 0, stream>>>(x, WINT, b_in, X0, Hf, Hb);

  for (int l = 0; l < 3; ++l){
    for (int kk = 0; kk < 3; ++kk){
      const float*  hin_f = (kk == 0) ? Hf : HLf;
      const __bf16* hin_b = (kk == 0) ? Hb : HLb;
      k_zero<<<NN * H / 256, 256, 0, stream>>>(SACC, NN * H);
      k_scat<<<NE / 8, 256, 0, stream>>>(hin_f, srcI, dstI, SACC);
      k_gru<<<NN / 16, 256, 0, stream>>>(SACC, DEG, hin_b, hin_f,
          GT + (size_t)(l * 3 + kk) * 384 * 128,
          WHHB + (size_t)l * 384 * 128,
          bih + l * 384, bhh + l * 384, HLf, HLb);
    }
    k_p12<<<NN / 16, 128, 0, stream>>>(Hb, BTP, P);
    k_zero<<<NN * H / 256, 256, 0, stream>>>(SACC, NN * H);
    k_attn<<<NE / 8, 256, 0, stream>>>(P, srcI, dstI, eattr, TAB, TAB2, Wa2, ba2, SACC);
    k_finish<<<NN * H / 256, 256, 0, stream>>>(HLf, SACC, DEG, Hf, Hb);
  }

  // readout
  k_pool<<<NN * 256 / 256, 256, 0, stream>>>(X0, Hf, batch, GSUM);
  k_head<<<1, 128, 0, stream>>>(GSUM, BCNT, W1, b1, gam, bet, W2, (float*)d_out);
}
